// LlamaStyleAttention_63900523430398
// MI455X (gfx1250) — matrix operand order repaired and checked by reading, not compiled, not run on hardware
//
#include <hip/hip_runtime.h>

// ---------------------------------------------------------------------------
// Llama-style attention forward for MI455X (gfx1250, wave32, WMMA).
// Pipeline: QKV GEMMs (bf16 WMMA) -> RoPE -> flash attention (bf16 WMMA,
// online softmax, async global->LDS K staging) -> output projection.
// ---------------------------------------------------------------------------

typedef unsigned int  u32;
typedef unsigned short u16;
typedef unsigned long long u64;
typedef __bf16 bf16x16 __attribute__((ext_vector_type(16)));
typedef float  v8f     __attribute__((ext_vector_type(8)));
typedef u32    v8u     __attribute__((ext_vector_type(8)));

#define B_  2
#define T_  2048
#define C_  2048
#define H_  16
#define D_  128
#define SCALE_ 0.08838834764831843f   // 1/sqrt(128)

__device__ __forceinline__ u16 f2bf(float f) {
  u32 u = __float_as_uint(f);
  u32 r = u + 0x7fffu + ((u >> 16) & 1u);   // round-to-nearest-even
  return (u16)(r >> 16);
}
__device__ __forceinline__ float bf2f(u16 h) {
  return __uint_as_float(((u32)h) << 16);
}
__device__ __forceinline__ u32 pack2(float lo, float hi) {
  return (u32)f2bf(lo) | ((u32)f2bf(hi) << 16);
}
// A-fragment k index for 16-bit 16x16x32 WMMA (contiguous order):
// VGPR i holds k pair {2i,2i+1} for lanes 0-15, +16 for lanes 16-31.
__device__ __forceinline__ int afk(int i, int lane) {
  int k = 2 * i;
  return k + ((lane & 16) ? 16 : 0);
}

__device__ __forceinline__ v8f wmma_bf16(v8u a, v8u b, v8f c) {
  return __builtin_amdgcn_wmma_f32_16x16x32_bf16(
      false, __builtin_bit_cast(bf16x16, a),
      false, __builtin_bit_cast(bf16x16, b),
      (short)0, c, false, false);
}

// ---------------------------------------------------------------------------
// GEMM: C[M=4096, N=2048] = A[4096, 2048] x Bw[2048, 2048]
// MODE 0: A = f32 (x), epilogue scatters bf16 into (B,H,T,D) layout (Q/K/V).
// MODE 1: A = bf16 (Y), epilogue writes f32 row-major (final output).
// Block tile: 128(M) x 128(N) x 32(K), 256 threads = 8 waves.
// Wave w computes rows [16w,16w+16) x 128 cols = 8 WMMA tiles per K-step
// (one A fragment reused across 8 B fragments).
// ---------------------------------------------------------------------------
#define GM 4096
#define GN 2048
#define GK 2048
#define AST 34   // LDS row stride (bf16 elems): 68B -> conflict-free frag loads
#define BST 34

template <int MODE>
__global__ __launch_bounds__(256) void gemm_bf16(const void* __restrict__ Ain,
                                                 const float* __restrict__ Bw,
                                                 void* __restrict__ Out) {
  __shared__ u16 As[128 * AST];
  __shared__ u16 Bs[128 * BST];

  const int tid  = threadIdx.x;
  const int lane = tid & 31;
  const int w    = tid >> 5;
  const int m0   = blockIdx.y * 128;
  const int n0   = blockIdx.x * 128;

  v8f cacc[8];
  #pragma unroll
  for (int f = 0; f < 8; f++)
    #pragma unroll
    for (int r = 0; r < 8; r++) cacc[f][r] = 0.0f;

  for (int kb = 0; kb < GK; kb += 32) {
    // ---- stage A tile (128x32) as bf16 ----
    if (MODE == 0) {
      const float* A = (const float*)Ain;
      const int tr  = tid >> 3;          // 0..31
      const int tk4 = (tid & 7) * 4;     // 0..28
      #pragma unroll
      for (int rr = 0; rr < 128; rr += 32) {
        const float4 av =
            *(const float4*)(A + (size_t)(m0 + tr + rr) * GK + kb + tk4);
        u32* dst = (u32*)&As[(tr + rr) * AST + tk4];
        dst[0] = pack2(av.x, av.y);
        dst[1] = pack2(av.z, av.w);
      }
    } else {
      const u16* A = (const u16*)Ain;
      #pragma unroll
      for (int j = 0; j < 2; j++) {
        const int id = tid + j * 256;    // 0..511
        const int r  = id >> 2;          // 0..127
        const int k8 = (id & 3) * 8;     // 0..24
        const uint4 v =
            *(const uint4*)(A + (size_t)(m0 + r) * GK + kb + k8);
        u32* dst = (u32*)&As[r * AST + k8];
        dst[0] = v.x; dst[1] = v.y; dst[2] = v.z; dst[3] = v.w;
      }
    }
    // ---- stage B tile (32x128) transposed -> Bs[n][k] ----
    {
      const int tn4 = (tid & 31) * 4;    // 0..124
      const int tk  = tid >> 5;          // 0..7
      #pragma unroll
      for (int kk = tk; kk < 32; kk += 8) {
        const float4 bv =
            *(const float4*)(Bw + (size_t)(kb + kk) * GN + n0 + tn4);
        Bs[(tn4 + 0) * BST + kk] = f2bf(bv.x);
        Bs[(tn4 + 1) * BST + kk] = f2bf(bv.y);
        Bs[(tn4 + 2) * BST + kk] = f2bf(bv.z);
        Bs[(tn4 + 3) * BST + kk] = f2bf(bv.w);
      }
    }
    __syncthreads();

    // ---- WMMA: 1 A-fragment reused across 8 B-fragments ----
    const int am = w * 16 + (lane & 15);
    v8u af;
    #pragma unroll
    for (int i = 0; i < 8; i++)
      af[i] = *(const u32*)&As[am * AST + afk(i, lane)];
    const int kb2 = (lane & 16) ? 16 : 0;
    #pragma unroll
    for (int f = 0; f < 8; f++) {
      v8u bfv;
      const int bn = f * 16 + (lane & 15);
      #pragma unroll
      for (int i = 0; i < 8; i++)
        bfv[i] = *(const u32*)&Bs[bn * BST + kb2 + 2 * i];
      cacc[f] = wmma_bf16(af, bfv, cacc[f]);
    }
    __syncthreads();
  }

  // ---- epilogue ----
  const int mlo = (lane & 16) ? 8 : 0;
  const int nn  = lane & 15;
  #pragma unroll
  for (int f = 0; f < 8; f++) {
    #pragma unroll
    for (int r = 0; r < 8; r++) {
      const int gm = m0 + w * 16 + r + mlo;
      const int gn = n0 + f * 16 + nn;
      if (MODE == 0) {
        // scatter into (B,H,T,D) bf16
        const int b = gm >> 11, t = gm & (T_ - 1);
        const int h = gn >> 7,  d = gn & (D_ - 1);
        ((u16*)Out)[(((size_t)(b * H_ + h) * T_ + t) * D_) + d] =
            f2bf(cacc[f][r]);
      } else {
        ((float*)Out)[(size_t)gm * GN + gn] = cacc[f][r];
      }
    }
  }
}

// ---------------------------------------------------------------------------
// RoPE on Q and K, layout (B,H,T,D), pairs (d, d+64). One thread per pair.
// ---------------------------------------------------------------------------
__global__ __launch_bounds__(256) void rope_qk(u16* __restrict__ Qb,
                                               u16* __restrict__ Kb) {
  const int idx = blockIdx.x * 256 + threadIdx.x;  // B*H*T*64 = 4194304
  const int d   = idx & 63;
  const int t   = (idx >> 6) & (T_ - 1);
  const int bh  = idx >> 17;
  const size_t base = ((size_t)bh * T_ + t) * D_;
  const float inv = __powf(10000.0f, -(float)d * (1.0f / 64.0f));
  float s, c;
  __sincosf((float)t * inv, &s, &c);

  {
    const float x0 = bf2f(Qb[base + d]), x1 = bf2f(Qb[base + d + 64]);
    Qb[base + d]      = f2bf(x0 * c - x1 * s);
    Qb[base + d + 64] = f2bf(x1 * c + x0 * s);
  }
  {
    const float x0 = bf2f(Kb[base + d]), x1 = bf2f(Kb[base + d + 64]);
    Kb[base + d]      = f2bf(x0 * c - x1 * s);
    Kb[base + d + 64] = f2bf(x1 * c + x0 * s);
  }
}

// ---------------------------------------------------------------------------
// Flash attention fwd. Block = 4 waves = 128 threads, one (b,h,q-block of 64).
// Each wave owns 16 q rows; loop over 32-key tiles (causal).
// K tile is staged with async global->LDS DMA (ASYNCcnt); V is transposed
// through VGPRs (DMA cannot transpose).
// ---------------------------------------------------------------------------
#define KSTR 136  // K tile LDS row stride (bf16): 272B -> conflict-free B-frag
#define VSTR 34
#define PSTR 34

__global__ __launch_bounds__(128) void flash_fwd(const u16* __restrict__ Qb,
                                                 const u16* __restrict__ Kb,
                                                 const u16* __restrict__ Vb,
                                                 u16* __restrict__ Yb) {
  __shared__ u16 Ks[32 * KSTR];        // [tk][d]
  __shared__ u16 Vs[128 * VSTR];       // [d][tk] (transposed)
  __shared__ u16 Ps[4 * 16 * PSTR];    // per-wave P bounce buffer

  const int tid  = threadIdx.x;
  const int lane = tid & 31;
  const int wave = tid >> 5;
  const int bh   = blockIdx.x >> 5;    // 32 q-blocks per head
  const int qblk = blockIdx.x & 31;
  const int q0   = qblk * 64;
  const int q0w  = q0 + wave * 16;
  const int b    = bh >> 4, h = bh & 15;

  const u16* Qh = Qb + (size_t)bh * T_ * D_;
  const u16* Kh = Kb + (size_t)bh * T_ * D_;
  const u16* Vh = Vb + (size_t)bh * T_ * D_;
  const u64 kbase64 = (u64)(size_t)Kh;

  // Q A-fragments (d split into 4 chunks of 32), loaded once.
  v8u qf[4];
  const int arow = q0w + (lane & 15);
  #pragma unroll
  for (int c = 0; c < 4; c++)
    #pragma unroll
    for (int i = 0; i < 8; i++)
      qf[c][i] = *(const u32*)(Qh + (size_t)arow * D_ + c * 32 + afk(i, lane));

  v8f acc[8];
  v8f mrow, lrow;
  #pragma unroll
  for (int r = 0; r < 8; r++) { mrow[r] = -3.0e38f; lrow[r] = 0.0f; }
  #pragma unroll
  for (int f = 0; f < 8; f++)
    #pragma unroll
    for (int r = 0; r < 8; r++) acc[f][r] = 0.0f;

  const int mlo = (lane & 16) ? 8 : 0;
  const int kb2 = (lane & 16) ? 16 : 0;
  const int ktiles = (q0 + 64) >> 5;
  // scalar (SGPR) copy of the wave's last q row for EXEC-preserving branches
  const int qlast_s = __builtin_amdgcn_readfirstlane(q0w + 15);

  // Per-thread fixed staging coordinates for the K tile (32x128 bf16).
  const int ktk = tid >> 4;              // 0..7   (row block, +8 per step)
  const int kdg = (tid & 15) * 8;        // 0..120 (d group of 8 bf16)
  const u32 klds0 = (u32)(size_t)&Ks[ktk * KSTR + kdg];

  for (int kt = 0; kt < ktiles; kt++) {
    const int k0 = kt * 32;
    if (kt + 1 < ktiles) {   // pull next tiles toward L2/WGP$
      __builtin_prefetch(Kh + (size_t)(k0 + 32) * D_, 0, 1);
      __builtin_prefetch(Vh + (size_t)(k0 + 32) * D_, 0, 1);
    }
    // ---- stage K tile (32x128) via async global->LDS DMA (ASYNCcnt) ----
    #pragma unroll
    for (int j = 0; j < 4; j++) {
      const u32 ldsoff = klds0 + (u32)(j * 8 * KSTR * 2);
      const u32 goff   = (u32)(((k0 + ktk + j * 8) * D_ + kdg) * 2);
      asm volatile("global_load_async_to_lds_b128 %0, %1, %2"
                   :: "v"(ldsoff), "v"(goff), "s"(kbase64)
                   : "memory");
    }
    // ---- stage V tile transposed (Vs[d][tk]) through VGPRs ----
    #pragma unroll
    for (int j = 0; j < 16; j++) {
      const int id = tid + j * 128;        // 0..2047
      const int tk = id >> 6, d2 = (id & 63) * 2;
      const u32 v = *(const u32*)(Vh + (size_t)(k0 + tk) * D_ + d2);
      Vs[(d2 + 0) * VSTR + tk] = (u16)(v & 0xffffu);
      Vs[(d2 + 1) * VSTR + tk] = (u16)(v >> 16);
    }
    asm volatile("s_wait_asynccnt 0" ::: "memory");
    __syncthreads();

    if (k0 <= qlast_s) {  // scalar branch: EXEC stays all-ones around WMMA
      // ---- scores = Q * K^T (16x32 = two 16x16 fragments) ----
      v8f sc[2];
      #pragma unroll
      for (int cf = 0; cf < 2; cf++)
        #pragma unroll
        for (int r = 0; r < 8; r++) sc[cf][r] = 0.0f;
      #pragma unroll
      for (int c = 0; c < 4; c++) {
        #pragma unroll
        for (int cf = 0; cf < 2; cf++) {
          v8u bfv;
          const int col = cf * 16 + (lane & 15);
          #pragma unroll
          for (int i = 0; i < 8; i++)
            bfv[i] = *(const u32*)&Ks[col * KSTR + c * 32 + kb2 + 2 * i];
          sc[cf] = wmma_bf16(qf[c], bfv, sc[cf]);
        }
      }
      // ---- scale + causal mask + tile row-max ----
      v8f tmax;
      #pragma unroll
      for (int r = 0; r < 8; r++) {
        const int qr = q0w + r + mlo;
        float s0 = sc[0][r] * SCALE_;
        float s1 = sc[1][r] * SCALE_;
        if (k0 + (lane & 15) > qr)      s0 = -1.0e30f;
        if (k0 + 16 + (lane & 15) > qr) s1 = -1.0e30f;
        sc[0][r] = s0; sc[1][r] = s1;
        tmax[r] = fmaxf(s0, s1);
      }
      #pragma unroll
      for (int off = 1; off < 16; off <<= 1)
        #pragma unroll
        for (int r = 0; r < 8; r++)
          tmax[r] = fmaxf(tmax[r], __shfl_xor(tmax[r], off, 32));
      // ---- online softmax update ----
      v8f alpha, rsum;
      #pragma unroll
      for (int r = 0; r < 8; r++) {
        const float mn = fmaxf(mrow[r], tmax[r]);
        alpha[r] = __expf(mrow[r] - mn);
        const float p0 = __expf(sc[0][r] - mn);
        const float p1 = __expf(sc[1][r] - mn);
        sc[0][r] = p0; sc[1][r] = p1;
        rsum[r] = p0 + p1;
        mrow[r] = mn;
      }
      #pragma unroll
      for (int off = 1; off < 16; off <<= 1)
        #pragma unroll
        for (int r = 0; r < 8; r++)
          rsum[r] += __shfl_xor(rsum[r], off, 32);
      #pragma unroll
      for (int r = 0; r < 8; r++) lrow[r] = lrow[r] * alpha[r] + rsum[r];
      #pragma unroll
      for (int f = 0; f < 8; f++)
        #pragma unroll
        for (int r = 0; r < 8; r++) acc[f][r] *= alpha[r];

      // ---- C-layout P -> LDS -> A-fragment (same-wave region) ----
      u16* Pw = &Ps[wave * 16 * PSTR];
      #pragma unroll
      for (int cf = 0; cf < 2; cf++)
        #pragma unroll
        for (int r = 0; r < 8; r++)
          Pw[(r + mlo) * PSTR + cf * 16 + (lane & 15)] = f2bf(sc[cf][r]);
      asm volatile("s_wait_dscnt 0" ::: "memory");  // same-wave LDS RAW
      v8u pa;
      const int pm = lane & 15;
      #pragma unroll
      for (int i = 0; i < 8; i++)
        pa[i] = *(const u32*)&Pw[pm * PSTR + afk(i, lane)];

      // ---- acc += P * V ----
      #pragma unroll
      for (int f = 0; f < 8; f++) {
        v8u vbv;
        const int col = f * 16 + (lane & 15);
        #pragma unroll
        for (int i = 0; i < 8; i++)
          vbv[i] = *(const u32*)&Vs[col * VSTR + kb2 + 2 * i];
        acc[f] = wmma_bf16(pa, vbv, acc[f]);
      }
    }
    __syncthreads();
  }

  // ---- epilogue: normalize and write Y (B,T,C) bf16 ----
  const int nn = lane & 15;
  #pragma unroll
  for (int r = 0; r < 8; r++) {
    const int qr = q0w + r + mlo;
    const float inv = 1.0f / lrow[r];
    #pragma unroll
    for (int f = 0; f < 8; f++)
      Yb[((size_t)(b * T_ + qr) * C_) + h * D_ + f * 16 + nn] =
          f2bf(acc[f][r] * inv);
  }
}

// ---------------------------------------------------------------------------
// Host-side launch. Workspace: Qb | Kb | Vb | Yb (bf16), 16 MB each = 64 MB.
// ---------------------------------------------------------------------------
extern "C" void kernel_launch(void* const* d_in, const int* in_sizes, int n_in,
                              void* d_out, int out_size, void* d_ws,
                              size_t ws_size, hipStream_t stream) {
  (void)in_sizes; (void)n_in; (void)out_size; (void)ws_size;
  const float* x  = (const float*)d_in[0];
  const float* wq = (const float*)d_in[1];
  const float* wk = (const float*)d_in[2];
  const float* wv = (const float*)d_in[3];
  const float* wo = (const float*)d_in[4];

  const size_t elems = (size_t)B_ * H_ * T_ * D_;  // 8M bf16 = 16 MB
  u16* Qb = (u16*)d_ws;
  u16* Kb = Qb + elems;
  u16* Vb = Kb + elems;
  u16* Yb = Vb + elems;

  dim3 gg(GN / 128, GM / 128);  // 16 x 32
  gemm_bf16<0><<<gg, 256, 0, stream>>>((const void*)x, wq, (void*)Qb);
  gemm_bf16<0><<<gg, 256, 0, stream>>>((const void*)x, wk, (void*)Kb);
  gemm_bf16<0><<<gg, 256, 0, stream>>>((const void*)x, wv, (void*)Vb);

  rope_qk<<<(B_ * H_ * T_ * 64) / 256, 256, 0, stream>>>(Qb, Kb);

  flash_fwd<<<B_ * H_ * (T_ / 64), 128, 0, stream>>>(Qb, Kb, Vb, Yb);

  gemm_bf16<1><<<gg, 256, 0, stream>>>((const void*)Yb, wo, d_out);
}
